// Seq2Seq_78116865180174
// MI455X (gfx1250) — matrix operand order repaired and checked by reading, not compiled, not run on hardware
//
#include <hip/hip_runtime.h>

#define VV 32000
#define EE 1024
#define HH 512
#define H2 1024
#define LL 2
#define BB 16
#define SS 128
#define TT 128

typedef __attribute__((ext_vector_type(16))) __bf16 v16bf;
typedef __attribute__((ext_vector_type(8)))  float  v8f;

union V8 { v8f v; float f[8]; };

__device__ __forceinline__ unsigned short f2bf(float x) {
  union { float f; unsigned u; } c; c.f = x;
  unsigned u = c.u + 0x7FFFu + ((c.u >> 16) & 1u);   // round-to-nearest-even
  return (unsigned short)(u >> 16);
}

__device__ __forceinline__ float sigmoidf_(float x) {
  return 1.0f / (1.0f + __expf(-x));
}

// ---- WMMA operand loaders (layouts per CDNA5 ISA 7.12.2) ----------------
// A: 16x32 bf16 tile. lane&15 = row M. lanes<16: K=k0+0..7 (v0-3), k0+16..23 (v4-7)
//                                      lanes>=16: K=k0+8..15,       k0+24..31
__device__ __forceinline__ v16bf load_a(const unsigned short* base, int lda, int k0) {
  int lane = threadIdx.x & 31;
  const unsigned short* p = base + (size_t)(lane & 15) * lda + k0 + ((lane >> 4) << 3);
  v16bf a;
  uint4* ap = reinterpret_cast<uint4*>(&a);
  ap[0] = *reinterpret_cast<const uint4*>(p);
  ap[1] = *reinterpret_cast<const uint4*>(p + 16);
  return a;
}

// B: 32x16 tile of W^T where W is [N,K] row-major. lane&15 = col N (= row of W).
// lanes<16 hold K=k0+0..7, k0+16..23; lanes>=16 hold K=k0+8..15, k0+24..31.
__device__ __forceinline__ v16bf load_b(const unsigned short* W, int ldw, int n0, int k0) {
  int lane = threadIdx.x & 31;
  const unsigned short* p = W + (size_t)(n0 + (lane & 15)) * ldw + k0 + ((lane >> 4) << 3);
  v16bf b;
  uint4* bp = reinterpret_cast<uint4*>(&b);
  bp[0] = reinterpret_cast<const uint4*>(p)[0];
  bp[1] = reinterpret_cast<const uint4*>(p + 16)[0];
  return b;
}

__device__ __forceinline__ v8f wmma_bf16(v16bf a, v16bf b, v8f c) {
  return __builtin_amdgcn_wmma_f32_16x16x32_bf16(false, a, false, b, (short)0, c,
                                                 false, false);
}

// ---- double-buffered 2x4-tile GEMM core ---------------------------------
// acc[2][4] += A(2 M-tiles) @ W^T(4 N-tiles) over K. Prefetches the next
// K=32 chunk's 6 operand tiles while the current 8 WMMAs execute.
__device__ __forceinline__ void gemm_core_2x4(const unsigned short* A0,
                                              const unsigned short* A1, int lda,
                                              const unsigned short* W, int ldw,
                                              int n00, int K, V8 acc[2][4]) {
  v16bf a0 = load_a(A0, lda, 0);
  v16bf a1 = load_a(A1, lda, 0);
  v16bf b0 = load_b(W, ldw, n00, 0);
  v16bf b1 = load_b(W, ldw, n00 + 16, 0);
  v16bf b2 = load_b(W, ldw, n00 + 32, 0);
  v16bf b3 = load_b(W, ldw, n00 + 48, 0);
  int k = 0;
  for (; k + 32 < K; k += 32) {
    int kn = k + 32;
    v16bf na0 = load_a(A0, lda, kn);
    v16bf na1 = load_a(A1, lda, kn);
    v16bf nb0 = load_b(W, ldw, n00, kn);
    v16bf nb1 = load_b(W, ldw, n00 + 16, kn);
    v16bf nb2 = load_b(W, ldw, n00 + 32, kn);
    v16bf nb3 = load_b(W, ldw, n00 + 48, kn);
    acc[0][0].v = wmma_bf16(a0, b0, acc[0][0].v);
    acc[1][0].v = wmma_bf16(a1, b0, acc[1][0].v);
    acc[0][1].v = wmma_bf16(a0, b1, acc[0][1].v);
    acc[1][1].v = wmma_bf16(a1, b1, acc[1][1].v);
    acc[0][2].v = wmma_bf16(a0, b2, acc[0][2].v);
    acc[1][2].v = wmma_bf16(a1, b2, acc[1][2].v);
    acc[0][3].v = wmma_bf16(a0, b3, acc[0][3].v);
    acc[1][3].v = wmma_bf16(a1, b3, acc[1][3].v);
    a0 = na0; a1 = na1; b0 = nb0; b1 = nb1; b2 = nb2; b3 = nb3;
  }
  acc[0][0].v = wmma_bf16(a0, b0, acc[0][0].v);
  acc[1][0].v = wmma_bf16(a1, b0, acc[1][0].v);
  acc[0][1].v = wmma_bf16(a0, b1, acc[0][1].v);
  acc[1][1].v = wmma_bf16(a1, b1, acc[1][1].v);
  acc[0][2].v = wmma_bf16(a0, b2, acc[0][2].v);
  acc[1][2].v = wmma_bf16(a1, b2, acc[1][2].v);
  acc[0][3].v = wmma_bf16(a0, b3, acc[0][3].v);
  acc[1][3].v = wmma_bf16(a1, b3, acc[1][3].v);
}

// ---- elementwise helpers -------------------------------------------------
__global__ void to_bf16_kernel(const float* in, unsigned short* out, long long n) {
  long long i = (long long)blockIdx.x * blockDim.x + threadIdx.x;
  long long stride = (long long)gridDim.x * blockDim.x;
  for (; i < n; i += stride) out[i] = f2bf(in[i]);
}

__global__ void embed_src_kernel(const int* src, const float* tab, unsigned short* X) {
  int row = blockIdx.x;                 // s*16 + b
  int s = row >> 4, b = row & 15;
  int tok = src[b * SS + s];
  const float* e = tab + (size_t)tok * EE;
  for (int j = threadIdx.x; j < EE; j += blockDim.x)
    X[(size_t)row * EE + j] = f2bf(e[j]);
}

__global__ void embed_dec_kernel(const int* tgt, const float* tab, unsigned short* X) {
  int row = blockIdx.x;                 // t*16 + b, t in 0..126
  int t = row >> 4, b = row & 15;
  int tok = (t == 0) ? 1 : tgt[b * TT + t];     // BOS = 1
  const float* e = tab + (size_t)tok * EE;
  for (int j = threadIdx.x; j < EE; j += blockDim.x)
    X[(size_t)row * EE + j] = f2bf(e[j]);
}

__global__ void zero_first_kernel(float* out) {  // out[:,0,:] = 0
  int i = blockIdx.x * blockDim.x + threadIdx.x;
  if (i < BB * VV) {
    int b = i / VV, v = i - b * VV;
    out[(size_t)b * TT * VV + v] = 0.0f;
  }
}

// ---- blocked GEMM: C[M,N] = A[M,K](bf16) @ W[N,K]^T (bf16) + bias --------
// __launch_bounds__(256, 1): min 1 wave/EU -> allocator may use the full
// per-wave VGPR budget so the 64-reg accumulator block + double buffer stays
// in registers (no scratch spills in the hot loop).
__global__ void __launch_bounds__(256, 1) gemm_kernel(
    const unsigned short* A, int lda, const unsigned short* W, int ldw,
    const float* bias, float* C, int ldc, int Mtiles, int Ntiles, int K) {
  int Mb = (Mtiles + 1) >> 1;
  int Nb = Ntiles >> 2;                           // Ntiles % 4 == 0 for all uses
  int wave = (int)((blockIdx.x * blockDim.x + threadIdx.x) >> 5);
  if (wave >= Mb * Nb) return;
  int mb = wave / Nb, nb = wave - mb * Nb;
  int mt0 = mb * 2, n00 = nb * 64;
  const unsigned short* A0 = A + (size_t)mt0 * 16 * lda;
  const unsigned short* A1 = A0 + (size_t)16 * lda;
  V8 acc[2][4];
#pragma unroll
  for (int i = 0; i < 2; ++i)
#pragma unroll
    for (int j = 0; j < 4; ++j) acc[i][j].v = (v8f){0, 0, 0, 0, 0, 0, 0, 0};
  gemm_core_2x4(A0, A1, lda, W, ldw, n00, K, acc);
  int lane = threadIdx.x & 31;
  int nl = lane & 15, rb = (lane >> 4) << 3;
#pragma unroll
  for (int i = 0; i < 2; ++i) {
    if (mt0 + i >= Mtiles) break;                 // wave-uniform guard
    int mbase = (mt0 + i) * 16 + rb;
#pragma unroll
    for (int j = 0; j < 4; ++j) {
      int n = n00 + j * 16 + nl;
      float bv = bias ? bias[n] : 0.0f;
#pragma unroll
      for (int g = 0; g < 8; ++g)
        C[(size_t)(mbase + g) * ldc + n] = acc[i][j].f[g] + bv;
    }
  }
}

// ---- blocked FC GEMM writing straight into d_out[b, t+1, v] --------------
__global__ void __launch_bounds__(256, 1) fc_kernel(
    const unsigned short* A, const unsigned short* W, const float* bias,
    float* out) {
  const int Nb = (VV / 16) / 4;                   // 500
  const int Mb = ((TT - 1) + 1) / 2;              // 64 (127 M-tiles, padded)
  int wave = (int)((blockIdx.x * blockDim.x + threadIdx.x) >> 5);
  if (wave >= Mb * Nb) return;
  int mb = wave / Nb, nb = wave - mb * Nb;
  int t0 = mb * 2, n00 = nb * 64;                 // M-tile index == timestep t
  const unsigned short* A0 = A + (size_t)t0 * 16 * H2;
  const unsigned short* A1 = A0 + (size_t)16 * H2;
  V8 acc[2][4];
#pragma unroll
  for (int i = 0; i < 2; ++i)
#pragma unroll
    for (int j = 0; j < 4; ++j) acc[i][j].v = (v8f){0, 0, 0, 0, 0, 0, 0, 0};
  gemm_core_2x4(A0, A1, H2, W, H2, n00, H2, acc);
  int lane = threadIdx.x & 31;
  int nl = lane & 15, rb = (lane >> 4) << 3;
#pragma unroll
  for (int i = 0; i < 2; ++i) {
    int t = t0 + i;
    if (t >= TT - 1) break;                       // wave-uniform guard
#pragma unroll
    for (int j = 0; j < 4; ++j) {
      int n = n00 + j * 16 + nl;
      float bv = bias[n];
#pragma unroll
      for (int g = 0; g < 8; ++g) {
        int b = rb + g;
        out[((size_t)b * TT + (t + 1)) * VV + n] = acc[i][j].f[g] + bv;
      }
    }
  }
}

// ---- encoder scan: one block per direction, persistent over S steps ------
// XW_base: [2][S][16][4H] precomputed x@Wih^T+b (f32). Whh_base: [2][4H][H] bf16.
// Y (bf16 [S][16][2H], concat fwd|bwd) may be null. h_out bf16 / c_out f32 [16][2H].
__global__ void __launch_bounds__(1024) enc_scan_kernel(
    const float* XW_base, const unsigned short* Whh_base,
    unsigned short* Y, unsigned short* h_out, float* c_out) {
  extern __shared__ unsigned short dyn_lds[];
  unsigned short* hbuf = dyn_lds;                 // [16][HH] bf16
  const int dir = blockIdx.x;
  const float* XW = XW_base + (size_t)dir * SS * 16 * (4 * HH);
  const unsigned short* Whh = Whh_base + (size_t)dir * (4 * HH) * HH;
  const int wave = threadIdx.x >> 5;              // 32 waves = 32 hidden tiles
  const int lane = threadIdx.x & 31;
  const int ht = wave;
  const int ncl = lane & 15;
  const int rb = (lane >> 4) << 3;
  for (int i = threadIdx.x; i < 16 * HH; i += blockDim.x) hbuf[i] = 0;
  float cst[8], hlast[8];
#pragma unroll
  for (int g = 0; g < 8; ++g) { cst[g] = 0.0f; hlast[g] = 0.0f; }
  __syncthreads();
  for (int s = 0; s < SS; ++s) {
    int srow = dir ? (SS - 1 - s) : s;
    const float* xw = XW + (size_t)srow * 16 * (4 * HH);
    V8 z[4];
#pragma unroll
    for (int gg = 0; gg < 4; ++gg) {
      int n = gg * HH + ht * 16 + ncl;
#pragma unroll
      for (int g = 0; g < 8; ++g) z[gg].f[g] = xw[(size_t)(rb + g) * (4 * HH) + n];
    }
    for (int k = 0; k < HH; k += 32) {
      v16bf a = load_a(hbuf, HH, k);
#pragma unroll
      for (int gg = 0; gg < 4; ++gg)
        z[gg].v = wmma_bf16(a, load_b(Whh, HH, gg * HH + ht * 16, k), z[gg].v);
    }
    __syncthreads();
#pragma unroll
    for (int g = 0; g < 8; ++g) {
      float iv = sigmoidf_(z[0].f[g]);
      float fv = sigmoidf_(z[1].f[g]);
      float gv = tanhf(z[2].f[g]);
      float ov = sigmoidf_(z[3].f[g]);
      float c2 = fv * cst[g] + iv * gv;
      float h2 = ov * tanhf(c2);
      cst[g] = c2; hlast[g] = h2;
      int m = rb + g;
      unsigned short hb = f2bf(h2);
      hbuf[m * HH + ht * 16 + ncl] = hb;
      if (Y) Y[((size_t)srow * 16 + m) * (2 * HH) + dir * HH + ht * 16 + ncl] = hb;
    }
    __syncthreads();
  }
#pragma unroll
  for (int g = 0; g < 8; ++g) {
    int m = rb + g;
    int col = dir * HH + ht * 16 + ncl;
    h_out[(size_t)m * H2 + col] = f2bf(hlast[g]);
    c_out[(size_t)m * H2 + col] = cst[g];
  }
}

// ---- decoder scan: single persistent block, 2 layers, 127 steps ----------
// Hidden tiles processed sequentially per wave (4 live accumulators, not 8)
// to stay inside the 8-waves/SIMD register budget of a 1024-thread block.
__global__ void __launch_bounds__(1024) dec_scan_kernel(
    const float* XW0, const unsigned short* Wih1, const unsigned short* Whh0,
    const unsigned short* Whh1, const float* b1,
    const unsigned short* h_init, const float* c_init, unsigned short* H1) {
  extern __shared__ unsigned short dyn_lds[];
  unsigned short* h0 = dyn_lds;                   // [16][H2] bf16
  unsigned short* h1 = dyn_lds + 16 * H2;         // [16][H2] bf16
  const int wave = threadIdx.x >> 5;
  const int lane = threadIdx.x & 31;
  const int ncl = lane & 15;
  const int rb = (lane >> 4) << 3;
  for (int i = threadIdx.x; i < 16 * H2; i += blockDim.x) {
    h0[i] = h_init[i];
    h1[i] = h_init[16 * H2 + i];
  }
  float c0[2][8], c1[2][8];
  float bini[2][4];                               // layer-1 bias, constant over t
#pragma unroll
  for (int hti = 0; hti < 2; ++hti) {
    int ht = wave + hti * 32;
#pragma unroll
    for (int g = 0; g < 8; ++g) {
      c0[hti][g] = c_init[(size_t)(rb + g) * H2 + ht * 16 + ncl];
      c1[hti][g] = c_init[16 * H2 + (size_t)(rb + g) * H2 + ht * 16 + ncl];
    }
#pragma unroll
    for (int gg = 0; gg < 4; ++gg) bini[hti][gg] = b1[gg * H2 + ht * 16 + ncl];
  }
  __syncthreads();
  for (int t = 0; t < TT - 1; ++t) {
    const float* xw = XW0 + (size_t)t * 16 * (4 * H2);
    float h2n[2][8];
    // ---------------- layer 0: z = xW+b (precomputed) + h0@Whh0^T ---------
#pragma unroll
    for (int hti = 0; hti < 2; ++hti) {
      int ht = wave + hti * 32;
      V8 z[4];
#pragma unroll
      for (int gg = 0; gg < 4; ++gg) {
        int n = gg * H2 + ht * 16 + ncl;
#pragma unroll
        for (int g = 0; g < 8; ++g)
          z[gg].f[g] = xw[(size_t)(rb + g) * (4 * H2) + n];
      }
      for (int k = 0; k < H2; k += 32) {
        v16bf a = load_a(h0, H2, k);
#pragma unroll
        for (int gg = 0; gg < 4; ++gg)
          z[gg].v = wmma_bf16(a, load_b(Whh0, H2, gg * H2 + ht * 16, k), z[gg].v);
      }
#pragma unroll
      for (int g = 0; g < 8; ++g) {
        float iv = sigmoidf_(z[0].f[g]);
        float fv = sigmoidf_(z[1].f[g]);
        float gv = tanhf(z[2].f[g]);
        float ov = sigmoidf_(z[3].f[g]);
        float c2 = fv * c0[hti][g] + iv * gv;
        c0[hti][g] = c2;
        h2n[hti][g] = ov * tanhf(c2);
      }
    }
    __syncthreads();                              // all waves done reading h0
#pragma unroll
    for (int hti = 0; hti < 2; ++hti) {
      int ht = wave + hti * 32;
#pragma unroll
      for (int g = 0; g < 8; ++g)
        h0[(rb + g) * H2 + ht * 16 + ncl] = f2bf(h2n[hti][g]);
    }
    __syncthreads();
    // ---------------- layer 1: z = b1 + h0@Wih1^T + h1@Whh1^T -------------
#pragma unroll
    for (int hti = 0; hti < 2; ++hti) {
      int ht = wave + hti * 32;
      V8 z[4];
#pragma unroll
      for (int gg = 0; gg < 4; ++gg) {
        float bv = bini[hti][gg];
#pragma unroll
        for (int g = 0; g < 8; ++g) z[gg].f[g] = bv;
      }
      for (int k = 0; k < H2; k += 32) {
        v16bf a0 = load_a(h0, H2, k);
        v16bf a1 = load_a(h1, H2, k);
#pragma unroll
        for (int gg = 0; gg < 4; ++gg) {
          int n0 = gg * H2 + ht * 16;
          z[gg].v = wmma_bf16(a0, load_b(Wih1, H2, n0, k), z[gg].v);
          z[gg].v = wmma_bf16(a1, load_b(Whh1, H2, n0, k), z[gg].v);
        }
      }
#pragma unroll
      for (int g = 0; g < 8; ++g) {
        float iv = sigmoidf_(z[0].f[g]);
        float fv = sigmoidf_(z[1].f[g]);
        float gv = tanhf(z[2].f[g]);
        float ov = sigmoidf_(z[3].f[g]);
        float c2 = fv * c1[hti][g] + iv * gv;
        c1[hti][g] = c2;
        h2n[hti][g] = ov * tanhf(c2);
      }
    }
    __syncthreads();                              // all waves done reading h1
#pragma unroll
    for (int hti = 0; hti < 2; ++hti) {
      int ht = wave + hti * 32;
#pragma unroll
      for (int g = 0; g < 8; ++g) {
        unsigned short hb = f2bf(h2n[hti][g]);
        h1[(rb + g) * H2 + ht * 16 + ncl] = hb;
        H1[((size_t)t * 16 + rb + g) * H2 + ht * 16 + ncl] = hb;
      }
    }
    __syncthreads();
  }
}

// -------------------------------------------------------------------------
extern "C" void kernel_launch(void* const* d_in, const int* in_sizes, int n_in,
                              void* d_out, int out_size, void* d_ws, size_t ws_size,
                              hipStream_t stream) {
  (void)in_sizes; (void)n_in; (void)out_size; (void)ws_size;
  const int*   src       = (const int*)d_in[0];
  const int*   tgt       = (const int*)d_in[1];
  const float* enc_embed = (const float*)d_in[3];
  const float* enc_Wih   = (const float*)d_in[4];
  const float* enc_Whh   = (const float*)d_in[5];
  const float* enc_b     = (const float*)d_in[6];
  const float* dec_embed = (const float*)d_in[7];
  const float* dec_Wih   = (const float*)d_in[8];
  const float* dec_Whh   = (const float*)d_in[9];
  const float* dec_b     = (const float*)d_in[10];
  const float* fc_W      = (const float*)d_in[11];
  const float* fc_b      = (const float*)d_in[12];
  float* out = (float*)d_out;

  char* ws = (char*)d_ws;
  size_t off = 0;
  auto alloc = [&](size_t bytes) -> void* {
    void* p = ws + off;
    off = (off + bytes + 255) & ~(size_t)255;
    return p;
  };
  unsigned short* encWihB = (unsigned short*)alloc((size_t)LL * 2 * 4 * HH * EE * 2);
  unsigned short* encWhhB = (unsigned short*)alloc((size_t)LL * 2 * 4 * HH * HH * 2);
  unsigned short* decWihB = (unsigned short*)alloc((size_t)LL * 4 * H2 * EE * 2);
  unsigned short* decWhhB = (unsigned short*)alloc((size_t)LL * 4 * H2 * H2 * 2);
  unsigned short* fcWB    = (unsigned short*)alloc((size_t)VV * H2 * 2);
  unsigned short* Xs      = (unsigned short*)alloc((size_t)SS * 16 * EE * 2);
  // +16 pad rows: blocked GEMM may speculatively read one extra M-tile
  unsigned short* Xd      = (unsigned short*)alloc(((size_t)(TT - 1) * 16 + 16) * EE * 2);
  unsigned short* Y0      = (unsigned short*)alloc((size_t)SS * 16 * H2 * 2);
  unsigned short* H1      = (unsigned short*)alloc(((size_t)(TT - 1) * 16 + 16) * H2 * 2);
  unsigned short* hinit   = (unsigned short*)alloc((size_t)LL * 16 * H2 * 2);
  float* XWenc = (float*)alloc((size_t)2 * SS * 16 * 4 * HH * 4);
  float* XW0d  = (float*)alloc((size_t)(TT - 1) * 16 * 4 * H2 * 4);
  float* cinit = (float*)alloc((size_t)LL * 16 * H2 * 4);

  // 1) weight conversion to bf16 (~125MB total: stays hot in 192MB L2)
  to_bf16_kernel<<<2048, 256, 0, stream>>>(enc_Wih, encWihB, (long long)LL * 2 * 4 * HH * EE);
  to_bf16_kernel<<<2048, 256, 0, stream>>>(enc_Whh, encWhhB, (long long)LL * 2 * 4 * HH * HH);
  to_bf16_kernel<<<2048, 256, 0, stream>>>(dec_Wih, decWihB, (long long)LL * 4 * H2 * EE);
  to_bf16_kernel<<<2048, 256, 0, stream>>>(dec_Whh, decWhhB, (long long)LL * 4 * H2 * H2);
  to_bf16_kernel<<<4096, 256, 0, stream>>>(fc_W, fcWB, (long long)VV * H2);

  // 2) embeddings (teacher forcing => decoder inputs known ahead of time)
  embed_src_kernel<<<SS * 16, 256, 0, stream>>>(src, enc_embed, Xs);
  embed_dec_kernel<<<(TT - 1) * 16, 256, 0, stream>>>(tgt, dec_embed, Xd);

  auto gemm = [&](const unsigned short* A, int lda, const unsigned short* W, int ldw,
                  const float* bias, float* C, int ldc, int Mt, int Nt, int K) {
    int waves = ((Mt + 1) / 2) * (Nt / 4);
    int blocks = (waves + 7) / 8;                 // 256 thr = 8 waves / block
    gemm_kernel<<<blocks, 256, 0, stream>>>(A, lda, W, ldw, bias, C, ldc, Mt, Nt, K);
  };

  // 3) encoder layer 0: input-side GEMMs + recurrent scan (both dirs)
  for (int d = 0; d < 2; ++d)
    gemm(Xs, EE, encWihB + (size_t)d * 4 * HH * EE, EE, enc_b + (size_t)d * 4 * HH,
         XWenc + (size_t)d * SS * 16 * 4 * HH, 4 * HH, SS, 4 * HH / 16, EE);
  enc_scan_kernel<<<2, 1024, 16 * HH * 2, stream>>>(XWenc, encWhhB, Y0, hinit, cinit);

  // 4) encoder layer 1 (input = concat fwd|bwd of layer 0); Y unused
  for (int d = 0; d < 2; ++d)
    gemm(Y0, H2, encWihB + (size_t)(2 + d) * 4 * HH * EE, EE,
         enc_b + (size_t)(2 + d) * 4 * HH,
         XWenc + (size_t)d * SS * 16 * 4 * HH, 4 * HH, SS, 4 * HH / 16, EE);
  enc_scan_kernel<<<2, 1024, 16 * HH * 2, stream>>>(
      XWenc, encWhhB + (size_t)2 * 4 * HH * HH, (unsigned short*)nullptr,
      hinit + 16 * H2, cinit + 16 * H2);

  // 5) decoder layer-0 input-side GEMM for all 127 steps at once
  gemm(Xd, EE, decWihB, EE, dec_b, XW0d, 4 * H2, TT - 1, 4 * H2 / 16, EE);

  // 6) persistent 2-layer decoder scan (stores layer-1 hiddens for FC)
  dec_scan_kernel<<<1, 1024, 2 * 16 * H2 * 2, stream>>>(
      XW0d, decWihB + (size_t)4 * H2 * EE, decWhhB, decWhhB + (size_t)4 * H2 * H2,
      dec_b + 4 * H2, hinit, cinit, H1);

  // 7) outputs: zero step 0, then one giant FC GEMM into d_out
  zero_first_kernel<<<(BB * VV + 255) / 256, 256, 0, stream>>>(out);
  {
    int waves = (((TT - 1) + 1) / 2) * ((VV / 16) / 4);
    int blocks = (waves + 7) / 8;
    fc_kernel<<<blocks, 256, 0, stream>>>(H1, fcWB, fc_b, out);
  }
}
